// QuantizedConv2d_10720238371570
// MI455X (gfx1250) — 8-bit matrix operand order repaired and checked by reading, compiled, not run on hardware
//
#include <hip/hip_runtime.h>

typedef int v8i __attribute__((ext_vector_type(8)));

#define HW_    16384
#define CIN_   64
#define COUT_  64
#define PIXBLK 128

static __device__ __forceinline__ unsigned pack4_lsb(unsigned a, unsigned b,
                                                     unsigned c, unsigned d) {
#if __has_builtin(__builtin_amdgcn_perm)
  const unsigned t0 = __builtin_amdgcn_perm(b, a, 0x0C0C0400u);
  const unsigned t1 = __builtin_amdgcn_perm(d, c, 0x0C0C0400u);
  return __builtin_amdgcn_perm(t1, t0, 0x05040100u);
#else
  return (a & 255u) | ((b & 255u) << 8) | ((c & 255u) << 16) | ((d & 255u) << 24);
#endif
}

static __device__ __forceinline__ int bytesum_acc(unsigned p, int acc) {
#if __has_builtin(__builtin_amdgcn_sad_u8)
  return (int)__builtin_amdgcn_sad_u8(p, 0u, (unsigned)acc);
#else
  return acc + (int)(p & 255u) + (int)((p >> 8) & 255u) +
         (int)((p >> 16) & 255u) + (int)((p >> 24) & 255u);
#endif
}

static __device__ __forceinline__ float clamp255(float f) {
#if __has_builtin(__builtin_amdgcn_fmed3f)
  return __builtin_amdgcn_fmed3f(f, 0.0f, 255.0f);
#else
  return fminf(fmaxf(f, 0.0f), 255.0f);
#endif
}

__global__ __launch_bounds__(256) void qconv1x1_iu8_wmma(
    const int* __restrict__ xq,
    const float* __restrict__ in_scale_p,
    const int* __restrict__ in_zp_p,
    const int* __restrict__ w8,
    const float* __restrict__ w_scale,
    const int* __restrict__ w_zp,
    const float* __restrict__ bias,
    const float* __restrict__ out_scale_p,
    const int* __restrict__ out_zp_p,
    float* __restrict__ out)
{
  __shared__ unsigned int wpack[COUT_ * 16];
  __shared__ float rs_l[COUT_];
  __shared__ float beta_l[COUT_];
  __shared__ float wzp_l[COUT_];
  __shared__ float kizp_l;

  const int t = threadIdx.x;

  #pragma unroll
  for (int j4 = 0; j4 < 4; ++j4) {
    const int j = t * 4 + j4;
    const int4 w4 = ((const int4*)w8)[j];
    wpack[j] = pack4_lsb((unsigned)w4.x, (unsigned)w4.y,
                         (unsigned)w4.z, (unsigned)w4.w);
  }
  __syncthreads();

  if (t < COUT_) {
    int rsum = 0;
    #pragma unroll
    for (int kg = 0; kg < 16; ++kg) {
      const unsigned int p = wpack[t * 16 + kg];
      rsum += (int)(signed char)(p) + (int)(signed char)(p >> 8) +
              (int)(signed char)(p >> 16) + (int)(signed char)(p >> 24);
    }
    const float is  = *in_scale_p;
    const float os  = *out_scale_p;
    const float izp = (float)(*in_zp_p);
    const float ozp = (float)(*out_zp_p);
    const float r   = is * w_scale[t] / os;
    rs_l[t]   = r;
    beta_l[t] = bias[t] / os + ozp - izp * (float)rsum * r;
    wzp_l[t]  = (float)w_zp[t];
    if (t == 0) kizp_l = (float)CIN_ * izp;
  }
  __syncthreads();

  const int lane = t & 31;
  const int wave = t >> 5;
  const int lh   = lane >> 4;
  const int ln   = lane & 15;
  const int pixbase = blockIdx.x * PIXBLK + wave * 16;
  const int bimg = pixbase / HW_;
  const int hw   = (pixbase % HW_) + ln;
  const int* xb  = xq + bimg * (CIN_ * HW_) + hw;

  v8i bfrag;
  int sx = 0;
  #pragma unroll
  for (int v = 0; v < 8; ++v) {
    const int kbase = ((v >> 2) * 32) + lh * 16 + ((v & 3) * 4);
    const unsigned a = (unsigned)xb[(kbase + 0) * HW_];
    const unsigned b = (unsigned)xb[(kbase + 1) * HW_];
    const unsigned c = (unsigned)xb[(kbase + 2) * HW_];
    const unsigned d = (unsigned)xb[(kbase + 3) * HW_];
    const unsigned p = pack4_lsb(a, b, c, d);
    bfrag[v] = (int)p;
    sx = bytesum_acc(p, sx);
  }
  sx += __shfl_xor(sx, 16);
  const float adjf = (float)sx - kizp_l;

  const v8i czero = {};
  #pragma unroll
  for (int mt = 0; mt < 4; ++mt) {
    v8i afrag;
    #pragma unroll
    for (int v = 0; v < 8; ++v) {
      const int v2 = v & 3;
      const int kbase = ((v >> 2) * 32) + lh * 16 + (v2 * 4);
      afrag[v] = (int)wpack[(mt * 16 + ln) * 16 + (kbase >> 2)];
    }
    v8i dacc = __builtin_amdgcn_wmma_i32_16x16x64_iu8(
         true, afrag,  false, bfrag, czero,
         false,  false);

    #pragma unroll
    for (int r = 0; r < 8; ++r) {
      const int o = mt * 16 + lh * 8 + r;
      const float conv = fmaf(-wzp_l[o], adjf, (float)dacc[r]);
      const float f = clamp255(rintf(fmaf(conv, rs_l[o], beta_l[o])));
      out[(bimg * COUT_ + o) * HW_ + hw] = f;
    }
  }
}

extern "C" void kernel_launch(void* const* d_in, const int* in_sizes, int n_in,
                              void* d_out, int out_size, void* d_ws, size_t ws_size,
                              hipStream_t stream) {
  const int*   xq  = (const int*)d_in[0];
  const float* isc = (const float*)d_in[1];
  const int*   izp = (const int*)d_in[2];
  const int*   w8  = (const int*)d_in[3];
  const float* wsc = (const float*)d_in[4];
  const int*   wzp = (const int*)d_in[5];
  const float* bs  = (const float*)d_in[6];
  const float* osc = (const float*)d_in[7];
  const int*   ozp = (const int*)d_in[8];

  const int total_pixels = 32 * HW_;
  dim3 grid(total_pixels / PIXBLK);
  qconv1x1_iu8_wmma<<<grid, 256, 0, stream>>>(
      xq, isc, izp, w8, wsc, wzp, bs, osc, ozp, (float*)d_out);
}
